// HetVolumeDecoder_16338055594516
// MI455X (gfx1250) — matrix operand order repaired and checked by reading, not compiled, not run on hardware
//
#include <hip/hip_runtime.h>
#include <hip/hip_bf16.h>
#include <math.h>

// ---------------------------------------------------------------------------
// Problem constants (from setup_inputs): B=8, LAT=128, HIDDEN=256, V=65536,
// D=128. factor = 0.5*D = 64.
// ---------------------------------------------------------------------------
#define BATCH   8
#define LAT     128
#define HID     256
#define NPTS    65536
#define DVOL    128
#define D3      (DVOL * DVOL * DVOL)   // 2097152 = 1<<21

typedef __attribute__((ext_vector_type(16))) __bf16        v16bf;
typedef __attribute__((ext_vector_type(8)))  float         v8f;
typedef __attribute__((ext_vector_type(2)))  float         v2f;
typedef __attribute__((ext_vector_type(4)))  unsigned int  v4u;
typedef __attribute__((ext_vector_type(8)))  int           v8i;
typedef __attribute__((ext_vector_type(4)))  int           v4i;

union FragBF {
    v16bf v;
    unsigned int u[8];
    unsigned short s[16];
};

__device__ __forceinline__ unsigned short f2bf(float f) {
    unsigned int u = __float_as_uint(f);
    u += 0x7FFFu + ((u >> 16) & 1u);     // round-to-nearest-even
    return (unsigned short)(u >> 16);
}
__device__ __forceinline__ float bf2f(unsigned short b) {
    return __uint_as_float(((unsigned int)b) << 16);
}

// ---------------------------------------------------------------------------
// Kernel 1: SIREN MLP with v_wmma_f32_16x16x32_bf16.
// One block, 512 threads = 16 waves. Wave w owns N-columns [16w, 16w+16).
// A (16xK bf16) staged in LDS; batch rows 8..15 are zero-padded.
//
// ISA 16-bit A 16x32 layout: lane L -> m = L&15, g = L>>4;
//   half h (0..15) -> k = kb + h + 8*g + (h>=8 ? 8 : 0)   (pairs contiguous)
// ISA 16-bit B 32x16 layout: lane L -> n = L&15; half h -> k same as A
// ISA 32-bit C/D 16x16 layout: lane L, vgpr r -> m = r + 8*(L>>4), n = L&15
// ---------------------------------------------------------------------------
__device__ __forceinline__ v8f wmma_tile_bf16(
    const unsigned short (*hb)[HID], const float* __restrict__ w,
    int K, int lane, int nBase)
{
    v8f acc = {0.f, 0.f, 0.f, 0.f, 0.f, 0.f, 0.f, 0.f};
    const int g = lane >> 4;
    const int m = lane & 15;
    const int nl = lane & 15;
    for (int kb = 0; kb < K; kb += 32) {
        FragBF a, b;
        #pragma unroll
        for (int hp = 0; hp < 8; ++hp) {
            int h0 = 2 * hp;
            int k = kb + h0 + 8 * g + ((h0 >= 8) ? 8 : 0);
            a.u[hp] = *(const unsigned int*)&hb[m][k];      // 2 bf16, 4B aligned
        }
        #pragma unroll
        for (int hh = 0; hh < 16; ++hh) {
            int k = kb + hh + 8 * g + ((hh >= 8) ? 8 : 0);
            b.s[hh] = f2bf(w[k * HID + nBase + nl]);        // fp32 weight -> bf16
        }
        acc = __builtin_amdgcn_wmma_f32_16x16x32_bf16(
            false, a.v, false, b.v, (short)0, acc, false, false);
    }
    return acc;
}

__global__ __launch_bounds__(512) void siren_mlp_kernel(
    const float* __restrict__ x,
    const float* __restrict__ w0, const float* __restrict__ b0,
    const float* __restrict__ w1, const float* __restrict__ b1,
    const float* __restrict__ w2, const float* __restrict__ b2,
    const float* __restrict__ w3, const float* __restrict__ b3,
    const float* __restrict__ w4, const float* __restrict__ b4,
    float* __restrict__ h_out)                   // [8][256] fp32
{
    __shared__ unsigned short hb[16][HID];       // bf16 activations (A matrix)

    const int tid  = threadIdx.x;
    const int wave = tid >> 5;
    const int lane = tid & 31;
    const int g    = lane >> 4;
    const int nl   = lane & 15;
    const int nBase = wave * 16;
    const int n     = nBase + nl;

    // zero LDS (covers pad rows and unused K for layer 0)
    for (int i = tid; i < 16 * HID / 2; i += 512)
        ((unsigned int*)hb)[i] = 0u;
    __syncthreads();
    // stage x[8][128] as bf16
    for (int i = tid; i < BATCH * LAT; i += 512) {
        int r = i >> 7, c = i & 127;
        hb[r][c] = f2bf(x[i]);
    }
    __syncthreads();

    // ---- layer 0: h = sin(30*(x @ w0 + b0)), K = 128 ----
    {
        v8f acc = wmma_tile_bf16(hb, w0, LAT, lane, nBase);
        float bn = b0[n];
        __syncthreads();                          // all A reads done
        #pragma unroll
        for (int r = 0; r < 8; ++r) {
            int mm = r + 8 * g;
            float s  = __sinf(30.0f * (acc[r] + bn));  // unconditional -> no exec split
            float nv = (mm < BATCH) ? s : 0.0f;        // v_cndmask
            hb[mm][n] = f2bf(nv);
        }
        __syncthreads();
    }

    // ---- layers 1..3: h = sin(h + h @ w + b), K = 256 ----
    const float* wlist[3] = {w1, w2, w3};
    const float* blist[3] = {b1, b2, b3};
    for (int L = 0; L < 3; ++L) {
        v8f acc = wmma_tile_bf16(hb, wlist[L], HID, lane, nBase);
        float bn = blist[L][n];
        __syncthreads();
        #pragma unroll
        for (int r = 0; r < 8; ++r) {
            int mm = r + 8 * g;
            float hold = bf2f(hb[mm][n]);         // residual input (own element)
            float s  = __sinf(hold + acc[r] + bn);
            float nv = (mm < BATCH) ? s : 0.0f;
            hb[mm][n] = f2bf(nv);
        }
        __syncthreads();
    }

    // ---- layer 4: h = h @ w4 + b4  -> fp32 out, rows 0..7 ----
    {
        v8f acc = wmma_tile_bf16(hb, w4, HID, lane, nBase);
        float bn = b4[n];
        if (g == 0) {                             // lanes 0..15 hold rows 0..7
            #pragma unroll
            for (int r = 0; r < 8; ++r)
                h_out[r * HID + n] = acc[r] + bn;
        }
    }
}

// ---------------------------------------------------------------------------
// Kernel 2: heads (fp32, HBM-bound: 268MB of wc/wv) + trilinear splat.
// 256 blocks x 256 threads; thread owns point v. Accumulators are packed
// float2 so the 32 FMAs/k lower to V_PK_FMA_F32 / dual-issue FMA (halves the
// VALU op count; keeps the kernel on the 23.3 TB/s HBM roof).
// h [8][256] (8KB) is staged into LDS by the Tensor Data Mover when the
// builtin is available (TENSORcnt + s_wait_tensorcnt), else by plain loads.
// ---------------------------------------------------------------------------
__global__ __launch_bounds__(256) void heads_splat_kernel(
    const float* __restrict__ h,        // [8][256]
    const float* __restrict__ wc,       // [256][196608]
    const float* __restrict__ bc,       // [196608]
    const float* __restrict__ wv,       // [256][65536]
    const float* __restrict__ bv,       // [65536]
    const float* __restrict__ coords0,  // [65536][3]
    const float* __restrict__ values0,  // [65536]
    float* __restrict__ grid)           // [8][128^3], pre-zeroed
{
    __shared__ float hs[BATCH][HID];
    const int tid = threadIdx.x;

#if __has_builtin(__builtin_amdgcn_tensor_load_to_lds) && \
    __has_builtin(__builtin_amdgcn_s_wait_tensorcnt)
    if (tid < 32) {                               // wave 0 only (TDM ignores EXEC)
        unsigned lds_off = (unsigned)(uintptr_t)&hs[0][0];
        unsigned long long ga = (unsigned long long)(uintptr_t)h;
        // D# group0: count=1 | lds_addr | global_addr[56:0] | type=2
        v4u g0 = { 1u,
                   lds_off,
                   (unsigned)(ga & 0xFFFFFFFFu),
                   (unsigned)((ga >> 32) & 0x01FFFFFFu) | 0x80000000u };
        // D# group1: 1 x 2048 tile of 4-byte elements, stride 2048
        v8i g1 = { (int)0x00020000,               // wg_mask=0, data_size=4B
                   (int)(2048u << 16),            // tensor_dim0[15:0] @ bits 63:48
                   (int)(1u << 16),               // tensor_dim0 hi=0 | tensor_dim1=1
                   (int)(2048u << 16),            // tensor_dim1 hi=0 | tile_dim0=2048
                   1,                             // tile_dim1=1, tile_dim2=0
                   2048,                          // tensor_dim0_stride lo32
                   (int)(2048u << 16),            // stride hi=0 | dim1_stride lo16
                   0 };                           // dim1_stride hi
        v4i z4 = {0, 0, 0, 0};                    // groups 2/3 unused (<=2D)
        v8i z8 = {0, 0, 0, 0, 0, 0, 0, 0};        // 6-arg form (this toolchain)
        __builtin_amdgcn_tensor_load_to_lds(g0, g1, z4, z4, z8, 0);
        __builtin_amdgcn_s_wait_tensorcnt(0);
    }
#else
    for (int i = tid; i < BATCH * HID; i += 256)
        hs[i >> 8][i & 255] = h[i];
#endif
    __syncthreads();

    const int v = blockIdx.x * 256 + tid;

    v2f a01[BATCH] = {};                          // (delta_x, delta_y)
    v2f a2v[BATCH] = {};                          // (delta_z, value)
    const float* wcp = wc + (size_t)3 * v;
    const float* wvp = wv + v;

    for (int k0 = 0; k0 < HID; k0 += 8) {
        // pull rows k0+16.. into L2 (speculative global_prefetch_b8)
        __builtin_prefetch(wcp + (size_t)16 * 3 * NPTS, 0, 1);
        #pragma unroll
        for (int kk = 0; kk < 8; ++kk) {
            float c0 = wcp[0], c1 = wcp[1], c2 = wcp[2];
            float wk = wvp[0];
            v2f c01 = {c0, c1};
            v2f c2v = {c2, wk};
            #pragma unroll
            for (int b = 0; b < BATCH; ++b) {
                float hbv = hs[b][k0 + kk];       // LDS broadcast
                v2f hb2 = {hbv, hbv};
                a01[b] += hb2 * c01;              // -> v_pk_fma_f32 / v_dual_fmac
                a2v[b] += hb2 * c2v;
            }
            wcp += (size_t)3 * NPTS;
            wvp += NPTS;
        }
    }

    const float bc0 = bc[3 * v], bc1 = bc[3 * v + 1], bc2 = bc[3 * v + 2];
    const float bvv = bv[v], v00 = values0[v];
    const float p0 = coords0[3 * v], p1 = coords0[3 * v + 1], p2 = coords0[3 * v + 2];
    const float factor = 0.5f * (float)DVOL;

    for (int b = 0; b < BATCH; ++b) {
        float val = v00 + a2v[b].y + bvv;
        val = val > 0.0f ? val : 0.0f;            // relu
        float cx = factor * (p0 + a01[b].x + bc0) + factor;
        float cy = factor * (p1 + a01[b].y + bc1) + factor;
        float cz = factor * (p2 + a2v[b].x + bc2) + factor;
        float flx = floorf(cx), fly = floorf(cy), flz = floorf(cz);
        int ix = (int)flx, iy = (int)fly, iz = (int)flz;
        float fx = cx - flx, fy = cy - fly, fz = cz - flz;
        float* gb = grid + ((size_t)b << 21);
        #pragma unroll
        for (int c = 0; c < 8; ++c) {
            int ox = c & 1, oy = (c >> 1) & 1, oz = (c >> 2) & 1;
            float w = (ox ? fx : 1.0f - fx) *
                      (oy ? fy : 1.0f - fy) *
                      (oz ? fz : 1.0f - fz);
            int X = ix + ox, Y = iy + oy, Z = iz + oz;
            if ((unsigned)X < (unsigned)DVOL && (unsigned)Y < (unsigned)DVOL &&
                (unsigned)Z < (unsigned)DVOL) {
                atomicAdd(&gb[((Z * DVOL) + Y) * DVOL + X], val * w);
            }
        }
    }
}

// ---------------------------------------------------------------------------
// Kernel 3: separable 7-tap Gaussian (std=1, radius 3), zero-padded SAME.
// STRIDE is compile-time (1 / 128 / 16384) so /% lower to shifts.
// Grids fit in the 192MB L2, so these passes are L2-resident.
// ---------------------------------------------------------------------------
template <int STRIDE>
__global__ __launch_bounds__(256) void blur_pass_kernel(
    const float* __restrict__ in, float* __restrict__ out, int n)
{
    // normalized exp(-t^2/2) taps, t = 0..3
    const float K0 = 0.39905028f;
    const float K1 = 0.24203622f;
    const float K2 = 0.05400558f;
    const float K3 = 0.00443305f;

    int i = blockIdx.x * 256 + threadIdx.x;
    if (i >= n) return;
    int axis = (i / STRIDE) & (DVOL - 1);
    float acc = K0 * in[i];
    if (axis >= 1) acc += K1 * in[i - STRIDE];
    if (axis <= DVOL - 2) acc += K1 * in[i + STRIDE];
    if (axis >= 2) acc += K2 * in[i - 2 * STRIDE];
    if (axis <= DVOL - 3) acc += K2 * in[i + 2 * STRIDE];
    if (axis >= 3) acc += K3 * in[i - 3 * STRIDE];
    if (axis <= DVOL - 4) acc += K3 * in[i + 3 * STRIDE];
    out[i] = acc;
}

// ---------------------------------------------------------------------------
extern "C" void kernel_launch(void* const* d_in, const int* in_sizes, int n_in,
                              void* d_out, int out_size, void* d_ws, size_t ws_size,
                              hipStream_t stream)
{
    const float* x       = (const float*)d_in[0];
    const float* w0      = (const float*)d_in[1];
    const float* b0      = (const float*)d_in[2];
    const float* w1      = (const float*)d_in[3];
    const float* b1      = (const float*)d_in[4];
    const float* w2      = (const float*)d_in[5];
    const float* b2      = (const float*)d_in[6];
    const float* w3      = (const float*)d_in[7];
    const float* b3      = (const float*)d_in[8];
    const float* w4      = (const float*)d_in[9];
    const float* b4      = (const float*)d_in[10];
    const float* wc      = (const float*)d_in[11];
    const float* bc      = (const float*)d_in[12];
    const float* wv      = (const float*)d_in[13];
    const float* bv      = (const float*)d_in[14];
    const float* coords0 = (const float*)d_in[15];
    const float* values0 = (const float*)d_in[16];
    // d_in[17] = volume_size (==128, compile-time here)

    // workspace: gridA (64MB) | gridB (64MB) | h (8KB)
    float* gridA = (float*)d_ws;
    float* gridB = gridA + (size_t)BATCH * D3;
    float* h_ws  = gridB + (size_t)BATCH * D3;

    (void)hipMemsetAsync(gridA, 0, (size_t)BATCH * D3 * sizeof(float), stream);

    siren_mlp_kernel<<<1, 512, 0, stream>>>(x, w0, b0, w1, b1, w2, b2,
                                            w3, b3, w4, b4, h_ws);

    heads_splat_kernel<<<NPTS / 256, 256, 0, stream>>>(
        h_ws, wc, bc, wv, bv, coords0, values0, gridA);

    const int n = BATCH * D3;
    const int blocks = (n + 255) / 256;
    blur_pass_kernel<1>            <<<blocks, 256, 0, stream>>>(gridA, gridB, n);
    blur_pass_kernel<DVOL>         <<<blocks, 256, 0, stream>>>(gridB, gridA, n);
    blur_pass_kernel<DVOL * DVOL>  <<<blocks, 256, 0, stream>>>(gridA, (float*)d_out, n);
}
